// LSTM_25555055412055
// MI455X (gfx1250) — matrix operand order repaired and checked by reading, not compiled, not run on hardware
//
#include <hip/hip_runtime.h>
#include <hip/hip_bf16.h>

// ---------------- constants ----------------
#define BATCH 1024
#define TSTEPS 64
#define FDIM 512
#define HDIM 1024
#define KDIM (FDIM + HDIM)   // 1536
#define NGATES (4 * HDIM)    // 4096
#define FORGET_BIAS 1.0f

typedef __attribute__((ext_vector_type(16))) __bf16       v16bf;
typedef __attribute__((ext_vector_type(8)))  float        v8f;
typedef __attribute__((ext_vector_type(4)))  unsigned int u32x4;
typedef __attribute__((ext_vector_type(8)))  unsigned int u32x8;

// ---------------- helpers ----------------
static __device__ __forceinline__ unsigned short f32_to_bf16(float f) {
    unsigned int u = __builtin_bit_cast(unsigned int, f);
    u += 0x7fffu + ((u >> 16) & 1u);   // round-to-nearest-even
    return (unsigned short)(u >> 16);
}

static __device__ __forceinline__ float sigf(float x) {
    return 1.0f / (1.0f + __expf(-x));
}

// Build a v16bf fragment from two 16-byte chunks.
static __device__ __forceinline__ v16bf frag_from(const unsigned short* p0,
                                                  const unsigned short* p1) {
    u32x4 lo = *(const u32x4*)p0;
    u32x4 hi = *(const u32x4*)p1;
    u32x8 t;
#pragma unroll
    for (int i = 0; i < 4; ++i) { t[i] = lo[i]; t[i + 4] = hi[i]; }
    return __builtin_bit_cast(v16bf, t);
}

// ---------------- prep kernels ----------------
__global__ void xcvt_kernel(const float4* __restrict__ x,
                            ushort4* __restrict__ xb, int n4) {
    int stride = gridDim.x * blockDim.x;
    for (int i = blockIdx.x * blockDim.x + threadIdx.x; i < n4; i += stride) {
        float4 v = x[i];
        ushort4 o;
        o.x = f32_to_bf16(v.x); o.y = f32_to_bf16(v.y);
        o.z = f32_to_bf16(v.z); o.w = f32_to_bf16(v.w);
        xb[i] = o;
    }
}

// W [KDIM][NGATES] fp32 row-major -> Wt [NGATES][KDIM] bf16 row-major
__global__ __launch_bounds__(256)
void wcvt_kernel(const float* __restrict__ W, unsigned short* __restrict__ Wt) {
    __shared__ float tile[32][33];
    int n0 = blockIdx.x * 32;   // column in W (row in Wt)
    int k0 = blockIdx.y * 32;   // row in W
    int tx = threadIdx.x & 31;
    int ty = threadIdx.x >> 5;  // 0..7
#pragma unroll
    for (int i = 0; i < 4; ++i) {
        int k = k0 + ty + i * 8;
        tile[ty + i * 8][tx] = W[(size_t)k * NGATES + n0 + tx];
    }
    __syncthreads();
#pragma unroll
    for (int i = 0; i < 4; ++i) {
        int n = n0 + ty + i * 8;
        Wt[(size_t)n * KDIM + k0 + tx] = f32_to_bf16(tile[tx][ty + i * 8]);
    }
}

__global__ void init_state_kernel(float* __restrict__ c,
                                  float* __restrict__ hf,
                                  unsigned short* __restrict__ ha,
                                  unsigned short* __restrict__ hb) {
    int i = blockIdx.x * blockDim.x + threadIdx.x;
    c[i] = 0.0f; hf[i] = 0.0f; ha[i] = 0; hb[i] = 0;
}

// ---------------- fused LSTM step: GEMM (WMMA bf16) + cell ----------------
// grid (16,16), block 256. Block computes rows m0..m0+63 x hidden cols n0..n0+63,
// for ALL FOUR gates, then applies the LSTM cell in registers.
__global__ __launch_bounds__(256)
void lstm_step_kernel(const unsigned short* __restrict__ xbf,  // [B,T,F] bf16
                      const unsigned short* __restrict__ Wt,   // [4H][K] bf16
                      const float* __restrict__ bias,          // [4H]
                      const unsigned short* __restrict__ h_in, // [B,H] bf16 (t-1)
                      unsigned short* __restrict__ h_out,      // [B,H] bf16 (t)
                      float* __restrict__ c_state,             // [B,H] f32
                      float* __restrict__ h_f32,               // [B,H] f32
                      int t) {
    __shared__ __align__(16) unsigned short ldsA[64 * 136];  // 64 rows x 128 halfs (+8 pad)

    const int tid = threadIdx.x;
    const int w   = tid >> 5;      // wave 0..7
    const int L   = tid & 31;
    const int mi  = w >> 1;        // m sub-tile 0..3
    const int nh  = w & 1;         // n half 0/1
    const int ln  = L & 15;
    const int kb2 = L >> 4;        // 0/1

    const int m0 = blockIdx.y * 64;
    const int n0 = blockIdx.x * 64;

    // B-matrix row pointers: lane = column n, K in 8-element groups (kb2 selects k 8..15 / 24..31).
    const unsigned short* brow[4][2];
#pragma unroll
    for (int g = 0; g < 4; ++g)
#pragma unroll
        for (int nj = 0; nj < 2; ++nj) {
            int ng = g * HDIM + n0 + nh * 32 + nj * 16 + ln;
            brow[g][nj] = Wt + (size_t)ng * KDIM + kb2 * 8;
        }

    const unsigned short* arow = &ldsA[(mi * 16 + ln) * 136];

    v8f acc[4][2];
#pragma unroll
    for (int g = 0; g < 4; ++g)
#pragma unroll
        for (int nj = 0; nj < 2; ++nj)
#pragma unroll
            for (int v = 0; v < 8; ++v) acc[g][nj][v] = 0.0f;

    for (int kc = 0; kc < 12; ++kc) {   // K chunks of 128
        __syncthreads();
        // stage A chunk: rows m0..m0+63, k = kc*128..+127 (x for kc<4, h for kc>=4)
#pragma unroll
        for (int i = 0; i < 4; ++i) {
            int c16 = tid + i * 256;
            int row = c16 >> 4;
            int col = (c16 & 15) << 3;   // half index
            int kg  = kc * 128 + col;
            const unsigned short* src = (kc < 4)
                ? xbf + ((size_t)(m0 + row) * TSTEPS + t) * FDIM + kg
                : h_in + (size_t)(m0 + row) * HDIM + (kg - FDIM);
            *(u32x4*)&ldsA[row * 136 + col] = *(const u32x4*)src;
        }
        __syncthreads();

#pragma unroll
        for (int ks = 0; ks < 4; ++ks) {
            const unsigned short* pa = arow + ks * 32 + kb2 * 8;
            v16bf af = frag_from(pa, pa + 16);
            int koff = kc * 128 + ks * 32;
#pragma unroll
            for (int g = 0; g < 4; ++g)
#pragma unroll
                for (int nj = 0; nj < 2; ++nj) {
                    const unsigned short* pb = brow[g][nj] + koff;
                    v16bf bf = frag_from(pb, pb + 16);
                    acc[g][nj] = __builtin_amdgcn_wmma_f32_16x16x32_bf16(
                        false, af, false, bf, (short)0, acc[g][nj], false, false);
                }
        }
    }

    // ---- LSTM cell, fully in registers (each lane owns its (m,n) cells) ----
#pragma unroll
    for (int nj = 0; nj < 2; ++nj) {
        int gn = n0 + nh * 32 + nj * 16 + ln;
        float bi = bias[0 * HDIM + gn];
        float bj = bias[1 * HDIM + gn];
        float bff = bias[2 * HDIM + gn];
        float bo = bias[3 * HDIM + gn];
#pragma unroll
        for (int v = 0; v < 8; ++v) {
            int gm = m0 + mi * 16 + v + 8 * kb2;   // C layout: M = v + 8*(lane>=16)
            size_t off = (size_t)gm * HDIM + gn;
            float iv = acc[0][nj][v] + bi;
            float jv = acc[1][nj][v] + bj;
            float fv = acc[2][nj][v] + bff;
            float ov = acc[3][nj][v] + bo;
            float cv = c_state[off];
            float nc = cv * sigf(fv + FORGET_BIAS) + sigf(iv) * tanhf(jv);
            float nh2 = tanhf(nc) * sigf(ov);
            c_state[off] = nc;
            h_f32[off]  = nh2;
            h_out[off]  = f32_to_bf16(nh2);
        }
    }
}

// ---------------- linear head: out[b] = h[b,:] . lw + lb ----------------
__global__ __launch_bounds__(256)
void head_kernel(const float* __restrict__ hf, const float* __restrict__ lw,
                 const float* __restrict__ lb, float* __restrict__ out) {
    int wv = threadIdx.x >> 5;
    int L  = threadIdx.x & 31;
    int b  = blockIdx.x * 8 + wv;
    const float* row = hf + (size_t)b * HDIM;
    float s = 0.0f;
    for (int k = L; k < HDIM; k += 32) s += row[k] * lw[k];
#pragma unroll
    for (int off = 16; off; off >>= 1) s += __shfl_xor(s, off, 32);
    if (L == 0) out[b] = s + lb[0];
}

// ---------------- launcher ----------------
extern "C" void kernel_launch(void* const* d_in, const int* in_sizes, int n_in,
                              void* d_out, int out_size, void* d_ws, size_t ws_size,
                              hipStream_t stream) {
    (void)in_sizes; (void)n_in; (void)out_size; (void)ws_size;
    const float* x  = (const float*)d_in[0];
    const float* W  = (const float*)d_in[1];
    const float* b  = (const float*)d_in[2];
    const float* lw = (const float*)d_in[3];
    const float* lb = (const float*)d_in[4];
    float* out = (float*)d_out;

    // workspace layout (256B aligned)
    unsigned char* ws = (unsigned char*)d_ws;
    size_t off = 0;
    auto take = [&](size_t bytes) {
        unsigned char* p = ws + off;
        off = (off + bytes + 255) & ~(size_t)255;
        return p;
    };
    unsigned short* Wt  = (unsigned short*)take((size_t)NGATES * KDIM * 2);          // 12.6 MB
    unsigned short* xbf = (unsigned short*)take((size_t)BATCH * TSTEPS * FDIM * 2);  // 64 MB
    unsigned short* hA  = (unsigned short*)take((size_t)BATCH * HDIM * 2);           // 2 MB
    unsigned short* hB  = (unsigned short*)take((size_t)BATCH * HDIM * 2);           // 2 MB
    float* hf32 = (float*)take((size_t)BATCH * HDIM * 4);                            // 4 MB
    float* cst  = (float*)take((size_t)BATCH * HDIM * 4);                            // 4 MB

    // 1) convert x -> bf16
    int n4 = (BATCH * TSTEPS * FDIM) / 4;
    xcvt_kernel<<<2048, 256, 0, stream>>>((const float4*)x, (ushort4*)xbf, n4);
    // 2) transpose + convert W -> Wt bf16 [4H][K]
    wcvt_kernel<<<dim3(NGATES / 32, KDIM / 32), 256, 0, stream>>>(W, Wt);
    // 3) zero state
    init_state_kernel<<<(BATCH * HDIM) / 256, 256, 0, stream>>>(cst, hf32, hA, hB);

    // 4) recurrence: one fused GEMM+cell launch per timestep (double-buffered h)
    for (int t = 0; t < TSTEPS; ++t) {
        const unsigned short* hin = (t & 1) ? hB : hA;
        unsigned short* hout      = (t & 1) ? hA : hB;
        lstm_step_kernel<<<dim3(16, 16), 256, 0, stream>>>(
            xbf, Wt, b, hin, hout, cst, hf32, t);
    }

    // 5) linear head on final h
    head_kernel<<<BATCH / 8, 256, 0, stream>>>(hf32, lw, lb, out);
}
